// GPT2Attention_26740466385787
// MI455X (gfx1250) — matrix operand order repaired and checked by reading, not compiled, not run on hardware
//
#include <hip/hip_runtime.h>
#include <hip/hip_bf16.h>

// ---------------------------------------------------------------------------
// GPT-2 attention, MI455X (gfx1250): bf16 WMMA + async global->LDS tile copies.
// ---------------------------------------------------------------------------

typedef __bf16 v16bf __attribute__((ext_vector_type(16)));
typedef __bf16 v8bf  __attribute__((ext_vector_type(8)));
typedef float  v8f   __attribute__((ext_vector_type(8)));

__device__ __forceinline__ unsigned short f2bf(float f) {
    unsigned u = __float_as_uint(f);
    unsigned r = u + 0x7fffu + ((u >> 16) & 1u);   // round-to-nearest-even
    return (unsigned short)(r >> 16);
}

__device__ __forceinline__ v8f wmma_bf16(v16bf a, v16bf b, v8f c) {
    return __builtin_amdgcn_wmma_f32_16x16x32_bf16(false, a, false, b,
                                                   (short)0, c, false, false);
}

// 16 contiguous bf16 (B fragments): lane supplies 16 K-values.
__device__ __forceinline__ v16bf ld_bfrag(const unsigned short* p) {
    v8bf lo = *(const v8bf*)(const void*)(p);
    v8bf hi = *(const v8bf*)(const void*)(p + 8);
    return __builtin_shufflevector(lo, hi, 0,1,2,3,4,5,6,7,8,9,10,11,12,13,14,15);
}

// A fragment (16-bit, 16x32): row pointer + kbase -> K kbase..kbase+7 and
// kbase+16..kbase+23 (ISA A-layout; kbase = (lane>=16)*8).
__device__ __forceinline__ v16bf ld_afrag(const unsigned short* row, int kbase) {
    v8bf lo = *(const v8bf*)(const void*)(row + kbase);
    v8bf hi = *(const v8bf*)(const void*)(row + kbase + 16);
    return __builtin_shufflevector(lo, hi, 0,1,2,3,4,5,6,7,8,9,10,11,12,13,14,15);
}

// CDNA5 async copy: 16 bytes global -> LDS, tracked by ASYNCcnt (no VGPRs).
__device__ __forceinline__ void async_cp16(unsigned lds_byte_addr,
                                           const unsigned short* g) {
    asm volatile("global_load_async_to_lds_b128 %0, %1, off"
                 :: "v"(lds_byte_addr), "v"(g) : "memory");
}
__device__ __forceinline__ void wait_async0() {
    asm volatile("s_wait_asynccnt 0x0" ::: "memory");
}
// Low 32 bits of a generic pointer to __shared__ == LDS byte offset.
__device__ __forceinline__ unsigned lds_addr(const void* p) {
    return (unsigned)(unsigned long long)p;
}

// --------------------------- conversion kernels ----------------------------

__global__ void cvt_f32_bf16_kernel(const float* __restrict__ in,
                                    unsigned short* __restrict__ out, int n) {
    int i = blockIdx.x * blockDim.x + threadIdx.x;
    if (i < n) out[i] = f2bf(in[i]);
}

// in[K][N] fp32 -> out[N][K] bf16 (coalesced writes)
__global__ void transpose_cvt_kernel(const float* __restrict__ in,
                                     unsigned short* __restrict__ out,
                                     int K, int N) {
    int i = blockIdx.x * blockDim.x + threadIdx.x;
    if (i < K * N) {
        int nn = i / K, kk = i - nn * K;
        out[i] = f2bf(in[(size_t)kk * N + nn]);
    }
}

// ------------------------------ GEMM kernel --------------------------------
// C[M,N] = A[M,K](bf16, row-major) * Bt[N,K](bf16)^T + bias
// Double-buffered LDS tiles filled with global_load_async_to_lds_b128.
// epi==0: fp32 store to outf;  epi==1: scatter into Q/K/Vt bf16 head layout.
#define LDK 40  // padded LDS row (32 data + 8 pad, keeps 16B alignment)

__global__ __launch_bounds__(256) void gemm_bf16_kernel(
    const unsigned short* __restrict__ A, const unsigned short* __restrict__ Bt,
    const float* __restrict__ bias, int M, int N, int K, int epi,
    unsigned short* __restrict__ qb, unsigned short* __restrict__ kb,
    unsigned short* __restrict__ vt, float* __restrict__ outf)
{
    __shared__ alignas(16) unsigned short As[2][128][LDK];
    __shared__ alignas(16) unsigned short Bs[2][128][LDK];

    const int t = threadIdx.x;
    const int lane = t & 31, wid = t >> 5;
    const int waveM = wid & 1, waveN = wid >> 1;       // 2 x 4 wave grid
    const int m0 = blockIdx.y * 128, n0 = blockIdx.x * 128;
    const int l16 = lane & 15, lh = lane >> 4;
    const int row = t >> 1, half = t & 1;              // tile-copy coords

    // Per-thread copy sources (advance by kt) and LDS destinations.
    const unsigned short* gA = A  + (size_t)(m0 + row) * K + half * 16;
    const unsigned short* gB = Bt + (size_t)(n0 + row) * K + half * 16;
    unsigned dA[2], dB[2];
    dA[0] = lds_addr(&As[0][row][half * 16]);
    dA[1] = lds_addr(&As[1][row][half * 16]);
    dB[0] = lds_addr(&Bs[0][row][half * 16]);
    dB[1] = lds_addr(&Bs[1][row][half * 16]);

    v8f acc[4][2];
#pragma unroll
    for (int i = 0; i < 4; ++i)
#pragma unroll
        for (int j = 0; j < 2; ++j)
#pragma unroll
            for (int e = 0; e < 8; ++e) acc[i][j][e] = 0.0f;

    const int nIter = K >> 5;  // K/32
    // Prologue: async-fill buffer 0 with tile 0.
    async_cp16(dA[0],      gA);
    async_cp16(dA[0] + 16, gA + 8);
    async_cp16(dB[0],      gB);
    async_cp16(dB[0] + 16, gB + 8);

    for (int it = 0; it < nIter; ++it) {
        const int cur = it & 1;
        wait_async0();          // my copies into buf[cur] complete
        __syncthreads();        // everyone's copies complete / reads drained

        if (it + 1 < nIter) {   // overlap next tile copy with compute
            const unsigned short* nA = gA + (it + 1) * 32;
            const unsigned short* nB = gB + (it + 1) * 32;
            const int nxt = cur ^ 1;
            async_cp16(dA[nxt],      nA);
            async_cp16(dA[nxt] + 16, nA + 8);
            async_cp16(dB[nxt],      nB);
            async_cp16(dB[nxt] + 16, nB + 8);
        }

        // Prefetch all fragments, then issue 8 back-to-back WMMAs.
        v16bf bf0 = ld_afrag(&Bs[cur][waveN * 32      + l16][0], lh * 8);
        v16bf bf1 = ld_afrag(&Bs[cur][waveN * 32 + 16 + l16][0], lh * 8);
        v16bf af[4];
#pragma unroll
        for (int mt = 0; mt < 4; ++mt)
            af[mt] = ld_afrag(&As[cur][waveM * 64 + mt * 16 + l16][0], lh * 8);
#pragma unroll
        for (int mt = 0; mt < 4; ++mt) {
            acc[mt][0] = wmma_bf16(af[mt], bf0, acc[mt][0]);
            acc[mt][1] = wmma_bf16(af[mt], bf1, acc[mt][1]);
        }
    }

#pragma unroll
    for (int mt = 0; mt < 4; ++mt)
#pragma unroll
        for (int nt = 0; nt < 2; ++nt) {
            v8f c = acc[mt][nt];
            int ccol = n0 + waveN * 32 + nt * 16 + l16;
            float bv = bias[ccol];
#pragma unroll
            for (int v = 0; v < 8; ++v) {
                int r = m0 + waveM * 64 + mt * 16 + v + 8 * lh;
                float val = c[v] + bv;
                if (epi == 0) {
                    outf[(size_t)r * N + ccol] = val;
                } else {
                    int sec = ccol >> 10, d = ccol & 1023;
                    int h = d >> 6, e = d & 63;
                    int b = r >> 11, s = r & 2047;
                    int bh = b * 16 + h;
                    unsigned short bfv = f2bf(val);
                    if (sec == 0)      qb[(size_t)(bh * 2048 + s) * 64 + e] = bfv;
                    else if (sec == 1) kb[(size_t)(bh * 2048 + s) * 64 + e] = bfv;
                    else               vt[(size_t)(bh * 64 + e) * 2048 + s] = bfv;
                }
            }
        }
}

// --------------------------- flash attention -------------------------------
// Grid: (S/64, B*H), 128 threads (4 wave32). Each wave owns 16 query rows.
// Q,K: [BH, S, 64] bf16; Vt: [BH, 64, S] bf16; AO: [B*S, 1024] bf16.
__global__ __launch_bounds__(128) void flash_attn_kernel(
    const unsigned short* __restrict__ Qb, const unsigned short* __restrict__ Kb,
    const unsigned short* __restrict__ Vt, unsigned short* __restrict__ AO)
{
    __shared__ alignas(16) unsigned short Plds[4][16][32];  // per-wave P tile

    const int t = threadIdx.x, lane = t & 31, w = t >> 5;
    const int l16 = lane & 15, lh = lane >> 4;
    const int bh = blockIdx.y;
    const int qbase = blockIdx.x * 64 + w * 16;

    const unsigned short* Qh = Qb + (size_t)bh * 2048 * 64;
    const unsigned short* Kh = Kb + (size_t)bh * 2048 * 64;
    const unsigned short* Vh = Vt + (size_t)bh * 64 * 2048;

    const int qrow = qbase + l16;
    v16bf aq0 = ld_afrag(Qh + (size_t)qrow * 64,      lh * 8);  // hd 0..31
    v16bf aq1 = ld_afrag(Qh + (size_t)qrow * 64 + 32, lh * 8);  // hd 32..63

    v8f o[4];
    float mrow[8], lrow[8];
#pragma unroll
    for (int nf = 0; nf < 4; ++nf)
#pragma unroll
        for (int e = 0; e < 8; ++e) o[nf][e] = 0.0f;
#pragma unroll
    for (int v = 0; v < 8; ++v) { mrow[v] = -__builtin_inff(); lrow[v] = 0.0f; }

    const int nkb = (qbase + 47) >> 5;   // causal: key blocks of 32
    for (int j = 0; j < nkb; ++j) {
        const int k0 = j * 32;

        // ---- S = Q * K^T (16q x 32k) ----
        v8f s[2];
#pragma unroll
        for (int kn = 0; kn < 2; ++kn) {
            const unsigned short* kp = Kh + (size_t)(k0 + kn * 16 + l16) * 64;
            v16bf b0 = ld_afrag(kp,      lh * 8);
            v16bf b1 = ld_afrag(kp + 32, lh * 8);
            v8f z;
#pragma unroll
            for (int e = 0; e < 8; ++e) z[e] = 0.0f;
            z = wmma_bf16(aq0, b0, z);
            z = wmma_bf16(aq1, b1, z);
            s[kn] = z;
        }

        // ---- causal mask + online softmax update ----
#pragma unroll
        for (int v = 0; v < 8; ++v) {
            int r = qbase + v + 8 * lh;
            float s0 = s[0][v] * 0.125f;           // 1/sqrt(64)
            float s1 = s[1][v] * 0.125f;
            if (k0 + l16      > r) s0 = -__builtin_inff();
            if (k0 + 16 + l16 > r) s1 = -__builtin_inff();
            s[0][v] = s0; s[1][v] = s1;
        }
        float alpha[8];
#pragma unroll
        for (int v = 0; v < 8; ++v) {
            float bm = fmaxf(s[0][v], s[1][v]);
#pragma unroll
            for (int off = 1; off < 16; off <<= 1)
                bm = fmaxf(bm, __shfl_xor(bm, off, 16));
            float mnew = fmaxf(mrow[v], bm);
            float p0 = __expf(s[0][v] - mnew);
            float p1 = __expf(s[1][v] - mnew);
            s[0][v] = p0; s[1][v] = p1;
            float rs = p0 + p1;
#pragma unroll
            for (int off = 1; off < 16; off <<= 1)
                rs += __shfl_xor(rs, off, 16);
            alpha[v] = __expf(mrow[v] - mnew);
            lrow[v]  = lrow[v] * alpha[v] + rs;
            mrow[v]  = mnew;
        }
#pragma unroll
        for (int nf = 0; nf < 4; ++nf)
#pragma unroll
            for (int v = 0; v < 8; ++v) o[nf][v] *= alpha[v];

        // ---- P (C-layout f32) -> LDS -> A-layout bf16 fragment ----
#pragma unroll
        for (int kn = 0; kn < 2; ++kn)
#pragma unroll
            for (int v = 0; v < 8; ++v)
                Plds[w][v + 8 * lh][kn * 16 + l16] = f2bf(s[kn][v]);
        asm volatile("s_wait_dscnt 0x0" ::: "memory");  // same-wave LDS RAW
        v16bf ap = ld_afrag(&Plds[w][l16][0], lh * 8);

        // ---- O += P * V  (V transposed: contiguous over keys) ----
#pragma unroll
        for (int nf = 0; nf < 4; ++nf) {
            const unsigned short* vp =
                Vh + (size_t)(nf * 16 + l16) * 2048 + k0;
            v16bf bv = ld_afrag(vp, lh * 8);
            o[nf] = wmma_bf16(ap, bv, o[nf]);
        }
    }

    // ---- normalize + store merged-head output (bf16) ----
    const int b = bh >> 4, h = bh & 15;
#pragma unroll
    for (int nf = 0; nf < 4; ++nf)
#pragma unroll
        for (int v = 0; v < 8; ++v) {
            int r = qbase + v + 8 * lh;
            float val = o[nf][v] / lrow[v];
            AO[((size_t)(b * 2048 + r)) * 1024 + h * 64 + nf * 16 + l16] = f2bf(val);
        }
}

// ------------------------------- launcher ----------------------------------

extern "C" void kernel_launch(void* const* d_in, const int* in_sizes, int n_in,
                              void* d_out, int out_size, void* d_ws, size_t ws_size,
                              hipStream_t stream) {
    const float* hidden = (const float*)d_in[0];   // [2,2048,1024]
    const float* W_attn = (const float*)d_in[1];   // [1024,3072]
    const float* b_attn = (const float*)d_in[2];   // [3072]
    const float* W_proj = (const float*)d_in[3];   // [1024,1024]
    const float* b_proj = (const float*)d_in[4];   // [1024]
    float* out = (float*)d_out;                    // [2,2048,1024] fp32

    const int B = 2, S = 2048, D = 1024, H = 16;
    const int M = B * S;          // 4096
    const size_t nX  = (size_t)M * D;        // 4194304
    const size_t nWA = (size_t)D * 3 * D;    // 3145728
    const size_t nWP = (size_t)D * D;        // 1048576

    unsigned short* ws = (unsigned short*)d_ws;
    unsigned short* Xb  = ws;                         // [4096,1024]
    unsigned short* WtA = Xb  + nX;                   // [3072,1024]
    unsigned short* WtP = WtA + nWA;                  // [1024,1024]
    unsigned short* Qb  = WtP + nWP;                  // [BH,S,64]
    unsigned short* Kb  = Qb  + nX;                   // [BH,S,64]
    unsigned short* Vt  = Kb  + nX;                   // [BH,64,S]
    unsigned short* AO  = Vt  + nX;                   // [4096,1024]

    // 1) fp32 -> bf16 conversions / weight transposes
    cvt_f32_bf16_kernel<<<(int)((nX + 255) / 256), 256, 0, stream>>>(hidden, Xb, (int)nX);
    transpose_cvt_kernel<<<(int)((nWA + 255) / 256), 256, 0, stream>>>(W_attn, WtA, D, 3 * D);
    transpose_cvt_kernel<<<(int)((nWP + 255) / 256), 256, 0, stream>>>(W_proj, WtP, D, D);

    // 2) QKV GEMM with per-head scatter epilogue (V stored transposed)
    gemm_bf16_kernel<<<dim3(3 * D / 128, M / 128), 256, 0, stream>>>(
        Xb, WtA, b_attn, M, 3 * D, D, /*epi=*/1, Qb, Kb, Vt, nullptr);

    // 3) flash attention: grid (q-blocks of 64, B*H heads)
    flash_attn_kernel<<<dim3(S / 64, B * H), 128, 0, stream>>>(Qb, Kb, Vt, AO);

    // 4) output projection -> fp32 result
    gemm_bf16_kernel<<<dim3(D / 128, M / 128), 256, 0, stream>>>(
        AO, WtP, b_proj, M, D, D, /*epi=*/0, nullptr, nullptr, nullptr, out);
}
